// Qwen3_5VisionModel_21955872817368
// MI455X (gfx1250) — compile-verified, matrix operand order checked by reading, not run on hardware
//
#include <hip/hip_runtime.h>
#include <hip/hip_bf16.h>

#define SEQ   2048
#define HID   1152
#define HEADS 16
#define HD    72
#define HDP   96      // padded head dim for QK^T (mult of 32)
#define HDV   80      // padded head dim for P@V N-dim (mult of 16)
#define ROTD  36
#define FF    4304
#define FFP   4320    // padded FF so fc2 K divides 32
#define SEG   512
#define NSEG  4
#define DEPTH 2
#define QKV3  (3*HID)

typedef __attribute__((ext_vector_type(16))) __bf16 v16bf;
typedef __attribute__((ext_vector_type(8)))  __bf16 v8bf;
typedef __attribute__((ext_vector_type(8)))  float  v8f;

__device__ __forceinline__ __bf16 to_bf16(float f){
  unsigned u = __builtin_bit_cast(unsigned, f);
  unsigned r = u + 0x7FFFu + ((u >> 16) & 1u);     // round-to-nearest-even
  unsigned short s = (unsigned short)(r >> 16);
  return __builtin_bit_cast(__bf16, s);
}

// A/B fragment for v_wmma_f32_16x16x32_bf16:
// lanes 0-15 hold K = k..k+7 (elems 0-7) and k+16..k+23 (elems 8-15);
// lanes 16-31 the same shifted by 8. Two b128 loads, K-contiguous source.
__device__ __forceinline__ v16bf load_frag(const __bf16* p){
  v8bf lo = *(const v8bf*)(p);
  v8bf hi = *(const v8bf*)(p + 16);
  v16bf r;
#pragma unroll
  for (int i = 0; i < 8; ++i){ r[i] = lo[i]; r[i + 8] = hi[i]; }
  return r;
}

// Generic batched GEMM: C[z][m][n] = sum_k A[z][m][k] * B[z][n][k]  (B row-major N x K)
// block = 4 waves; wave tile = 32(M) x 64(N); block tile 128 x 64.
// Two-deep software pipeline (double-buffered fragments), branch-free k-steps.
// mode 0: f32 C = acc + bias
// mode 1: f32 C += acc + bias            (residual)
// mode 2: bf16 C = gelu_tanh(acc + bias) (col < Nvalid)
// mode 3: f32 C = acc * scale
// mode 4: bf16 C = acc                   (col < Nvalid)
__global__ __launch_bounds__(128) void gemm_bf16_wmma(
    const __bf16* __restrict__ A, long long sA, int lda,
    const __bf16* __restrict__ B, long long sB, int ldb,
    void* __restrict__ Cv, long long sC, int ldc,
    const float* __restrict__ bias,
    int M, int N, int Nvalid, int K, float scale, int mode)
{
  const int wave = threadIdx.x >> 5;
  const int lane = threadIdx.x & 31;
  const int z    = blockIdx.z;
  A += (size_t)z * sA;
  B += (size_t)z * sB;

  const int m0 = blockIdx.x * 128 + wave * 32;
  const int n0 = blockIdx.y * 64;

  const int half = lane >> 4;
  const int ln   = lane & 15;
  const int koff = half * 8;

  const __bf16* Ar0 = A + (size_t)(m0 + ln) * lda + koff;
  const __bf16* Ar1 = Ar0 + (size_t)16 * lda;

  // Clamped per-tile B row pointers (branch-free inner loop)
  const __bf16* Br[4];
#pragma unroll
  for (int t = 0; t < 4; ++t){
    int nt = n0 + t * 16;
    if (nt > N - 16) nt = N - 16;
    Br[t] = B + (size_t)(nt + ln) * ldb + koff;
  }

  v8f acc[2][4] = {};
  v16bf a0[2], a1[2], b0[4], b1[4];
  const int kiter = K >> 5;

  // prologue: buffer 0 <- k = 0
  a0[0] = load_frag(Ar0);
  a0[1] = load_frag(Ar1);
#pragma unroll
  for (int t = 0; t < 4; ++t) b0[t] = load_frag(Br[t]);

  int k = 32;
#pragma unroll 1
  for (int it = 0; it + 2 <= kiter; it += 2){
    // prefetch buffer 1 <- k
    a1[0] = load_frag(Ar0 + k);
    a1[1] = load_frag(Ar1 + k);
#pragma unroll
    for (int t = 0; t < 4; ++t) b1[t] = load_frag(Br[t] + k);
    // compute with buffer 0 (k-32)
#pragma unroll
    for (int mi = 0; mi < 2; ++mi)
#pragma unroll
      for (int t = 0; t < 4; ++t)
        acc[mi][t] = __builtin_amdgcn_wmma_f32_16x16x32_bf16(
            false, a0[mi], false, b0[t], (short)0, acc[mi][t], false, false);
    // prefetch buffer 0 <- k+32 (clamped: redundant reload on final pass)
    const int k2 = (it + 2 < kiter) ? (k + 32) : k;
    a0[0] = load_frag(Ar0 + k2);
    a0[1] = load_frag(Ar1 + k2);
#pragma unroll
    for (int t = 0; t < 4; ++t) b0[t] = load_frag(Br[t] + k2);
    // compute with buffer 1 (k)
#pragma unroll
    for (int mi = 0; mi < 2; ++mi)
#pragma unroll
      for (int t = 0; t < 4; ++t)
        acc[mi][t] = __builtin_amdgcn_wmma_f32_16x16x32_bf16(
            false, a1[mi], false, b1[t], (short)0, acc[mi][t], false, false);
    k += 64;
  }
  if (kiter & 1){
    // tail k-step lives in buffer 0
#pragma unroll
    for (int mi = 0; mi < 2; ++mi)
#pragma unroll
      for (int t = 0; t < 4; ++t)
        acc[mi][t] = __builtin_amdgcn_wmma_f32_16x16x32_bf16(
            false, a0[mi], false, b0[t], (short)0, acc[mi][t], false, false);
  }

  // C layout: lane = col within tile; VGPR r -> row r (lanes<16) / r+8 (lanes>=16)
#pragma unroll
  for (int mi = 0; mi < 2; ++mi){
    const int rbase = m0 + mi * 16 + half * 8;
#pragma unroll
    for (int t = 0; t < 4; ++t){
      if (n0 + t * 16 >= N) continue;    // duplicated (clamped) tiles: skip store
      const int col = n0 + t * 16 + ln;
      const float bv = bias ? bias[col] : 0.0f;
#pragma unroll
      for (int r = 0; r < 8; ++r){
        const int rr = rbase + r;
        const float v = acc[mi][t][r];
        const size_t idx = (size_t)rr * ldc + col + (size_t)z * sC;
        if (mode == 0){
          ((float*)Cv)[idx] = v + bv;
        } else if (mode == 1){
          ((float*)Cv)[idx] += v + bv;
        } else if (mode == 2){
          if (col < Nvalid){
            float x = v + bv;
            float g = 0.5f * x * (1.0f + tanhf(0.7978845608028654f * (x + 0.044715f * x * x * x)));
            ((__bf16*)Cv)[idx] = to_bf16(g);
          }
        } else if (mode == 3){
          ((float*)Cv)[idx] = v * scale;
        } else {
          if (col < Nvalid) ((__bf16*)Cv)[idx] = to_bf16(v);
        }
      }
    }
  }
}

__global__ __launch_bounds__(256) void layernorm_to_bf16(
    const float* __restrict__ x, const float* __restrict__ w,
    const float* __restrict__ b, __bf16* __restrict__ out)
{
  __shared__ float red[256];
  const int row = blockIdx.x;
  const int tid = threadIdx.x;
  const float* xr = x + (size_t)row * HID;
  float s = 0.f;
  for (int i = tid; i < HID; i += 256) s += xr[i];
  red[tid] = s; __syncthreads();
  for (int o = 128; o > 0; o >>= 1){ if (tid < o) red[tid] += red[tid + o]; __syncthreads(); }
  const float mean = red[0] * (1.0f / HID);
  __syncthreads();
  float v = 0.f;
  for (int i = tid; i < HID; i += 256){ float dd = xr[i] - mean; v += dd * dd; }
  red[tid] = v; __syncthreads();
  for (int o = 128; o > 0; o >>= 1){ if (tid < o) red[tid] += red[tid + o]; __syncthreads(); }
  const float rs = rsqrtf(red[0] * (1.0f / HID) + 1e-6f);
  for (int i = tid; i < HID; i += 256)
    out[(size_t)row * HID + i] = to_bf16((xr[i] - mean) * rs * w[i] + b[i]);
}

// qkv f32 (S,3,HEADS,HD) -> rotary(q),rotary(k) into padded bf16 (HEADS,S,HDP),
// v into transposed padded bf16 (HEADS,HDV,S). Pads written zero every call.
__global__ __launch_bounds__(256) void rotary_split(
    const float* __restrict__ qkv, const float* __restrict__ rope,
    __bf16* __restrict__ qb, __bf16* __restrict__ kb, __bf16* __restrict__ vt)
{
  const int s = blockIdx.x;
  const __bf16 z0 = to_bf16(0.f);
  for (int i = threadIdx.x; i < HEADS * HDP; i += 256){
    const int h = i / HDP, d = i % HDP;
    const size_t qi = ((size_t)h * SEQ + s) * HDP + d;
    if (d >= HD){
      qb[qi] = z0; kb[qi] = z0;
      if (d < HDV) vt[((size_t)h * HDV + d) * SEQ + s] = z0;
      continue;
    }
    const int dm = (d < ROTD) ? d : d - ROTD;
    const float fr = rope[(size_t)s * ROTD + dm];
    const float c = cosf(fr), sn = sinf(fr);
    const float* base = qkv + (size_t)s * QKV3 + h * HD;
    const int dp = (d < ROTD) ? d + ROTD : d - ROTD;

    const float qv = base[d];
    const float qr = (d < ROTD) ? -base[dp] : base[dp];
    qb[qi] = to_bf16(qv * c + qr * sn);

    const float kv = base[HID + d];
    const float kr = (d < ROTD) ? -base[HID + dp] : base[HID + dp];
    kb[qi] = to_bf16(kv * c + kr * sn);

    vt[((size_t)h * HDV + d) * SEQ + s] = to_bf16(base[2 * HID + d]);
  }
}

// One wave per (head,row): softmax over the 512 in-segment scores.
__global__ __launch_bounds__(32) void softmax_rows(
    const float* __restrict__ sc, __bf16* __restrict__ P)
{
  const int row = blockIdx.x, h = blockIdx.y, lane = threadIdx.x;
  const float* sr = sc + ((size_t)h * SEG + row) * SEG;
  __bf16* pr = P + ((size_t)h * SEG + row) * SEG;
  float vals[16];
  float m = -1e30f;
#pragma unroll
  for (int j = 0; j < 16; ++j){ vals[j] = sr[lane + j * 32]; m = fmaxf(m, vals[j]); }
  for (int o = 16; o > 0; o >>= 1) m = fmaxf(m, __shfl_xor(m, o));
  float sum = 0.f;
#pragma unroll
  for (int j = 0; j < 16; ++j){ vals[j] = __expf(vals[j] - m); sum += vals[j]; }
  for (int o = 16; o > 0; o >>= 1) sum += __shfl_xor(sum, o);
  const float inv = 1.0f / sum;
#pragma unroll
  for (int j = 0; j < 16; ++j) pr[lane + j * 32] = to_bf16(vals[j] * inv);
}

__global__ void cvt_f32_bf16(const float* __restrict__ in, __bf16* __restrict__ out, long long n){
  long long i = (long long)blockIdx.x * blockDim.x + threadIdx.x;
  if (i < n) out[i] = to_bf16(in[i]);
}
__global__ void cvt_f32_bf16_pad(const float* __restrict__ in, __bf16* __restrict__ out,
                                 int rows, int cols, int ldo){
  long long i = (long long)blockIdx.x * blockDim.x + threadIdx.x;
  if (i >= (long long)rows * ldo) return;
  int r = (int)(i / ldo), c = (int)(i % ldo);
  out[i] = (c < cols) ? to_bf16(in[(size_t)r * cols + c]) : to_bf16(0.f);
}
__global__ void zero_ffpad(__bf16* __restrict__ ff){
  int i = blockIdx.x * blockDim.x + threadIdx.x;   // SEQ * 16
  int r = i >> 4, c = (i & 15) + FF;
  ff[(size_t)r * FFP + c] = to_bf16(0.f);
}

extern "C" void kernel_launch(void* const* d_in, const int* in_sizes, int n_in,
                              void* d_out, int out_size, void* d_ws, size_t ws_size,
                              hipStream_t stream) {
  const float* x_in    = (const float*)d_in[0];
  const float* rope    = (const float*)d_in[2];
  const float* norm1_w = (const float*)d_in[3];
  const float* norm1_b = (const float*)d_in[4];
  const float* qkv_w   = (const float*)d_in[5];
  const float* qkv_b   = (const float*)d_in[6];
  const float* proj_w  = (const float*)d_in[7];
  const float* proj_b  = (const float*)d_in[8];
  const float* norm2_w = (const float*)d_in[9];
  const float* norm2_b = (const float*)d_in[10];
  const float* fc1_w   = (const float*)d_in[11];
  const float* fc1_b   = (const float*)d_in[12];
  const float* fc2_w   = (const float*)d_in[13];
  const float* fc2_b   = (const float*)d_in[14];
  float* xout = (float*)d_out;

  char* ws = (char*)d_ws;
  size_t off = 0;
  auto alloc = [&](size_t bytes) -> char* {
    char* p = ws + off;
    off = (off + bytes + 255) & ~(size_t)255;
    return p;
  };
  __bf16* wqkv  = (__bf16*)alloc((size_t)DEPTH * QKV3 * HID * 2);
  __bf16* wproj = (__bf16*)alloc((size_t)DEPTH * HID * HID * 2);
  __bf16* wfc1  = (__bf16*)alloc((size_t)DEPTH * FF * HID * 2);
  __bf16* wfc2  = (__bf16*)alloc((size_t)DEPTH * HID * FFP * 2);
  __bf16* hbuf  = (__bf16*)alloc((size_t)SEQ * HID * 2);
  float*  qkvf  = (float*) alloc((size_t)SEQ * QKV3 * 4);
  __bf16* qb    = (__bf16*)alloc((size_t)HEADS * SEQ * HDP * 2);
  __bf16* kb    = (__bf16*)alloc((size_t)HEADS * SEQ * HDP * 2);
  __bf16* vt    = (__bf16*)alloc((size_t)HEADS * HDV * SEQ * 2);
  float*  sc    = (float*) alloc((size_t)HEADS * SEG * SEG * 4);
  __bf16* pb    = (__bf16*)alloc((size_t)HEADS * SEG * SEG * 2);
  __bf16* attnb = (__bf16*)alloc((size_t)SEQ * HID * 2);
  __bf16* ffb   = (__bf16*)alloc((size_t)SEQ * FFP * 2);

  // --- one-time per call: bf16 weight conversion + pad init ---
  {
    long long n;
    n = (long long)DEPTH * QKV3 * HID;
    cvt_f32_bf16<<<(unsigned)((n + 255) / 256), 256, 0, stream>>>(qkv_w, wqkv, n);
    n = (long long)DEPTH * HID * HID;
    cvt_f32_bf16<<<(unsigned)((n + 255) / 256), 256, 0, stream>>>(proj_w, wproj, n);
    n = (long long)DEPTH * FF * HID;
    cvt_f32_bf16<<<(unsigned)((n + 255) / 256), 256, 0, stream>>>(fc1_w, wfc1, n);
    n = (long long)DEPTH * HID * FFP;
    cvt_f32_bf16_pad<<<(unsigned)((n + 255) / 256), 256, 0, stream>>>(fc2_w, wfc2, DEPTH * HID, FF, FFP);
    zero_ffpad<<<(SEQ * 16) / 256, 256, 0, stream>>>(ffb);
  }
  hipMemcpyAsync(xout, x_in, (size_t)SEQ * HID * 4, hipMemcpyDeviceToDevice, stream);

  const float scale = 0.11785113019775793f;  // 1/sqrt(72)

  for (int d = 0; d < DEPTH; ++d){
    // LN1 -> bf16
    layernorm_to_bf16<<<SEQ, 256, 0, stream>>>(xout, norm1_w + d * HID, norm1_b + d * HID, hbuf);
    // qkv = h @ Wqkv^T + b   (f32 out)
    gemm_bf16_wmma<<<dim3(SEQ/128, QKV3/64, 1), 128, 0, stream>>>(
        hbuf, 0, HID, wqkv + (size_t)d * QKV3 * HID, 0, HID,
        qkvf, 0, QKV3, qkv_b + (size_t)d * QKV3, SEQ, QKV3, QKV3, HID, 1.f, 0);
    // rotary + split into padded bf16 Q/K and transposed V
    rotary_split<<<SEQ, 256, 0, stream>>>(qkvf, rope, qb, kb, vt);

    // block-diagonal attention: 4 segments of 512
    for (int seg = 0; seg < NSEG; ++seg){
      // scores = Q K^T * scale   (batched over heads)
      gemm_bf16_wmma<<<dim3(SEG/128, SEG/64, HEADS), 128, 0, stream>>>(
          qb + (size_t)seg * SEG * HDP, (long long)SEQ * HDP, HDP,
          kb + (size_t)seg * SEG * HDP, (long long)SEQ * HDP, HDP,
          sc, (long long)SEG * SEG, SEG, nullptr, SEG, SEG, SEG, HDP, scale, 3);
      softmax_rows<<<dim3(SEG, HEADS), 32, 0, stream>>>(sc, pb);
      // attn = P @ V  -> bf16 (S,HID), cols h*72..h*72+71
      gemm_bf16_wmma<<<dim3(SEG/128, (HDV + 63)/64, HEADS), 128, 0, stream>>>(
          pb, (long long)SEG * SEG, SEG,
          vt + (size_t)seg * SEG, (long long)HDV * SEQ, SEQ,
          attnb + (size_t)seg * SEG * HID, HD, HID, nullptr, SEG, HDV, HD, SEG, 1.f, 4);
    }
    // x += attn @ Wproj^T + b
    gemm_bf16_wmma<<<dim3(SEQ/128, HID/64, 1), 128, 0, stream>>>(
        attnb, 0, HID, wproj + (size_t)d * HID * HID, 0, HID,
        xout, 0, HID, proj_b + (size_t)d * HID, SEQ, HID, HID, HID, 1.f, 1);
    // LN2 -> bf16
    layernorm_to_bf16<<<SEQ, 256, 0, stream>>>(xout, norm2_w + d * HID, norm2_b + d * HID, hbuf);
    // ff = gelu(h @ Wfc1^T + b) -> bf16 (stride FFP, pad cols stay 0)
    gemm_bf16_wmma<<<dim3(SEQ/128, (FF + 63)/64, 1), 128, 0, stream>>>(
        hbuf, 0, HID, wfc1 + (size_t)d * FF * HID, 0, HID,
        ffb, 0, FFP, fc1_b + (size_t)d * FF, SEQ, FF, FF, HID, 1.f, 2);
    // x += ff @ Wfc2^T + b   (K = FFP, pad contributes 0)
    gemm_bf16_wmma<<<dim3(SEQ/128, HID/64, 1), 128, 0, stream>>>(
        ffb, 0, FFP, wfc2 + (size_t)d * HID * FFP, 0, FFP,
        xout, 0, HID, fc2_b + (size_t)d * HID, SEQ, HID, HID, FFP, 1.f, 1);
  }
  (void)in_sizes; (void)n_in; (void)out_size; (void)ws_size;
}
